// Subgraph_GNN_70961449664572
// MI455X (gfx1250) — matrix operand order repaired and checked by reading, not compiled, not run on hardware
//
#include <hip/hip_runtime.h>

typedef __attribute__((ext_vector_type(16))) __bf16        bf16x16;
typedef __attribute__((ext_vector_type(8)))  __bf16        bf16x8;
typedef __attribute__((ext_vector_type(8)))  float          v8f;
typedef __attribute__((ext_vector_type(8)))  float          f32x8;
typedef __attribute__((ext_vector_type(16))) unsigned short u16x16;
typedef __attribute__((ext_vector_type(8)))  unsigned short u16x8;

#define NROWS 8192
#define H1 128
#define KHALF (NROWS / 2)
#define PFD 1024  // prefetch distance in floats (4 KB, ~32 iterations ahead)

// round-to-nearest-even fp32 -> bf16 (scalar fallback)
__device__ __forceinline__ unsigned short f2bf(float f) {
  unsigned int u = __builtin_bit_cast(unsigned int, f);
  u += 0x7FFFu + ((u >> 16) & 1u);
  return (unsigned short)(u >> 16);
}

// packed fp32x8 -> bf16x8 (backend emits v_cvt_pk_bf16_f32)
__device__ __forceinline__ bf16x8 cvt8(f32x8 v) {
  return __builtin_convertvector(v, bf16x8);
}

__device__ __forceinline__ bf16x16 join16(bf16x8 lo, bf16x8 hi) {
  return __builtin_shufflevector(lo, hi, 0, 1, 2, 3, 4, 5, 6, 7, 8, 9, 10, 11,
                                 12, 13, 14, 15);
}

__device__ __forceinline__ v8f wmma_bf16(bf16x16 a, bf16x16 b, v8f c) {
  return __builtin_amdgcn_wmma_f32_16x16x32_bf16(false, a, false, b, (short)0, c,
                                                 false, false);
}

// ---------------------------------------------------------------------------
// prep_x: X = concat(x, walk_feats) -> bf16, stored (a) feature-major Xt[64][8192]
// for B operands, (b) row-major hops0[8192][64] (branch-0 input to the MLP).
// ---------------------------------------------------------------------------
__global__ void prep_x(const float* __restrict__ x, const float* __restrict__ wf,
                       unsigned short* __restrict__ Xt,
                       unsigned short* __restrict__ hops0) {
  int idx = blockIdx.x * blockDim.x + threadIdx.x;  // 0 .. 8192*64-1
  int n = idx >> 6;
  int f = idx & 63;
  float v = (f < 48) ? x[(size_t)n * 48 + f] : wf[(size_t)n * 16 + (f - 48)];
  unsigned short b = f2bf(v);
  Xt[(size_t)f * NROWS + n] = b;      // feature-major (K contiguous per feature)
  hops0[(size_t)n * 64 + f] = b;      // row-major
}

// ---------------------------------------------------------------------------
// prep_w: transpose weights to output-major bf16 so WMMA B columns are
// contiguous in K.  W1t[k][o(128)][i(64)], W2t[k][o(64)][i(128)].
// ---------------------------------------------------------------------------
__global__ void prep_w(const float* __restrict__ W1, const float* __restrict__ W2,
                       unsigned short* __restrict__ W1t,
                       unsigned short* __restrict__ W2t) {
  int idx = blockIdx.x * blockDim.x + threadIdx.x;  // 0 .. 32767
  {
    int k = idx >> 13, r = idx & 8191;
    int o = r >> 6, i = r & 63;                    // o:0..127, i:0..63
    W1t[idx] = f2bf(W1[(size_t)k * 8192 + (size_t)i * 128 + o]);
  }
  {
    int k = idx >> 13, r = idx & 8191;
    int o = r >> 7, i = r & 127;                   // o:0..63, i:0..127
    W2t[idx] = f2bf(W2[(size_t)k * 8192 + (size_t)i * 64 + o]);
  }
}

// ---------------------------------------------------------------------------
// hop_gemm: partial[b][kh] = hop_b[:, khalf] @ X[khalf].
// 3072 wave-tasks = 3 hops x 512 row tiles x 2 K-halves (K-split for 2x more
// latency-hiding waves on this HBM-streaming kernel; f32 partials, no atomics).
// Each wave: 16 output rows x 64 features (4 WMMA C tiles), K = 4096.
// Loop bounds are compile-time constants (kbeg folded into pointers) so loop
// control stays scalar; prefetch handled by a main/tail loop split.
// ---------------------------------------------------------------------------
__global__ void __launch_bounds__(256, 1)
hop_gemm(const float* __restrict__ hop1, const float* __restrict__ hop2,
         const float* __restrict__ hop3, const unsigned short* __restrict__ Xt,
         float* __restrict__ hopsP) {
  const int lane = threadIdx.x & 31;
  const int lcol = lane & 15;
  const int half = lane >> 4;
  const int wtask = blockIdx.x * 8 + (threadIdx.x >> 5);  // 0..3071
  const int branch = wtask >> 10;                         // 0..2
  const int rem = wtask & 1023;
  const int tile = rem >> 1;                              // 0..511
  const int kh = rem & 1;                                 // K half

  const float* hop = (branch == 0) ? hop1 : ((branch == 1) ? hop2 : hop3);
  const int row = tile * 16 + lcol;                       // A row for this lane
  const int kbeg = kh * KHALF;
  const float* arow = hop + (size_t)row * NROWS + kbeg + half * 16;
  const unsigned short* bbase = Xt + (size_t)lcol * NROWS + kbeg + half * 16;

  v8f acc[4] = {v8f{}, v8f{}, v8f{}, v8f{}};

  auto step = [&](int kk, bool pf) {
    // A fragment: lane = row (lcol), 16 contiguous floats k = 16*half + s
    f32x8 a0 = *(const f32x8*)(arow + kk);
    f32x8 a1 = *(const f32x8*)(arow + kk + 8);
    if (pf) __builtin_prefetch(arow + kk + PFD, 0, 3);
    bf16x16 a = join16(cvt8(a0), cvt8(a1));
#pragma unroll
    for (int t = 0; t < 4; ++t) {
      u16x16 bu = *(const u16x16*)(bbase + (size_t)t * 16 * NROWS + kk);
      acc[t] = wmma_bf16(a, __builtin_bit_cast(bf16x16, bu), acc[t]);
    }
  };

#pragma unroll 2
  for (int kk = 0; kk < KHALF - PFD; kk += 32) step(kk, true);
#pragma unroll 2
  for (int kk = KHALF - PFD; kk < KHALF; kk += 32) step(kk, false);

  // store f32 partial tiles: hopsP[(branch*2+kh)][row][feat]
  float* op = hopsP + ((size_t)(branch * 2 + kh) * NROWS + (size_t)tile * 16) * 64;
#pragma unroll
  for (int t = 0; t < 4; ++t) {
#pragma unroll
    for (int r = 0; r < 8; ++r) {
      int m = r + half * 8;  // C layout: row = r + 8*(lane/16), col = lane%16
      op[(size_t)m * 64 + t * 16 + lcol] = acc[t][r];
    }
  }
}

// ---------------------------------------------------------------------------
// mlp: per 16-row tile, 4 branches of (64 -> relu(128) -> 64), summed.
// Branch 0 reads bf16 X; branches 1-3 sum the two f32 K-partials and convert.
// Layer-1 output staged per-wave in LDS to flip C-layout -> A-layout.
// ---------------------------------------------------------------------------
__global__ void __launch_bounds__(256, 1)
mlp(const unsigned short* __restrict__ hops0, const float* __restrict__ hopsP,
    const unsigned short* __restrict__ W1t, const float* __restrict__ b1,
    const unsigned short* __restrict__ W2t, const float* __restrict__ b2,
    float* __restrict__ out) {
  __shared__ __align__(32) unsigned short Hs[8][16 * H1];  // 4 KB per wave

  const int lane = threadIdx.x & 31;
  const int lcol = lane & 15;
  const int half = lane >> 4;
  const int w = threadIdx.x >> 5;
  const int tile = blockIdx.x * 8 + w;  // 0..511

  v8f acc[4] = {v8f{}, v8f{}, v8f{}, v8f{}};

  for (int k = 0; k < 4; ++k) {
    // ---- layer 1 A fragments: tile [16 x 64] of branch-k input ----
    bf16x16 a1f[2];
    if (k == 0) {
      const unsigned short* Abase =
          hops0 + ((size_t)tile * 16 + lcol) * 64 + half * 16;
#pragma unroll
      for (int s = 0; s < 2; ++s) {
        u16x8 c0 = *(const u16x8*)(Abase + s * 32);
        u16x8 c1 = *(const u16x8*)(Abase + s * 32 + 8);
        a1f[s] = join16(__builtin_bit_cast(bf16x8, c0),
                        __builtin_bit_cast(bf16x8, c1));
      }
    } else {
      const float* P0 = hopsP +
          ((size_t)(k - 1) * 2 * NROWS + (size_t)tile * 16 + lcol) * 64 + half * 16;
      const float* P1 = P0 + (size_t)NROWS * 64;
#pragma unroll
      for (int s = 0; s < 2; ++s) {
        f32x8 x0 = *(const f32x8*)(P0 + s * 32) + *(const f32x8*)(P1 + s * 32);
        f32x8 x1 = *(const f32x8*)(P0 + s * 32 + 8) +
                   *(const f32x8*)(P1 + s * 32 + 8);
        a1f[s] = join16(cvt8(x0), cvt8(x1));
      }
    }

    __syncthreads();  // previous branch's Hs reads complete before overwrite
#pragma unroll
    for (int t = 0; t < 8; ++t) {
      int ocol = t * 16 + lcol;
      const unsigned short* Bb = W1t + ((size_t)k * H1 + ocol) * 64 + half * 16;
      v8f c = {};
      c = wmma_bf16(a1f[0], __builtin_bit_cast(bf16x16, *(const u16x16*)Bb), c);
      c = wmma_bf16(a1f[1], __builtin_bit_cast(bf16x16, *(const u16x16*)(Bb + 32)), c);
      float bv = b1[k * H1 + ocol];
      f32x8 h;
#pragma unroll
      for (int r = 0; r < 8; ++r) {
        float hv = c[r] + bv;
        h[r] = hv > 0.f ? hv : 0.f;
      }
      bf16x8 hb = cvt8(h);
#pragma unroll
      for (int r = 0; r < 8; ++r)
        Hs[w][(r + half * 8) * H1 + ocol] = __builtin_bit_cast(unsigned short, hb[r]);
    }
    __syncthreads();

    // ---- layer 2: A = H [16 x 128] from LDS, accumulate into acc ----
#pragma unroll
    for (int s = 0; s < 4; ++s) {
      const unsigned short* Ab = &Hs[w][lcol * H1 + s * 32 + half * 16];
      u16x8 c0 = *(const u16x8*)Ab;
      u16x8 c1 = *(const u16x8*)(Ab + 8);
      bf16x16 a = join16(__builtin_bit_cast(bf16x8, c0),
                         __builtin_bit_cast(bf16x8, c1));
#pragma unroll
      for (int t = 0; t < 4; ++t) {
        int ocol = t * 16 + lcol;
        const unsigned short* Bb =
            W2t + ((size_t)k * 64 + ocol) * H1 + s * 32 + half * 16;
        acc[t] = wmma_bf16(a, __builtin_bit_cast(bf16x16, *(const u16x16*)Bb),
                           acc[t]);
      }
    }
  }

  // store out[tile*16 .. +15][64] with sum_k b2[k] added
  float* op = out + (size_t)tile * 16 * 64;
#pragma unroll
  for (int t = 0; t < 4; ++t) {
    int ocol = t * 16 + lcol;
    float bsum = b2[ocol] + b2[64 + ocol] + b2[128 + ocol] + b2[192 + ocol];
#pragma unroll
    for (int r = 0; r < 8; ++r) {
      int m = r + half * 8;
      op[(size_t)m * 64 + ocol] = acc[t][r] + bsum;
    }
  }
}

// ---------------------------------------------------------------------------
extern "C" void kernel_launch(void* const* d_in, const int* in_sizes, int n_in,
                              void* d_out, int out_size, void* d_ws, size_t ws_size,
                              hipStream_t stream) {
  const float* x    = (const float*)d_in[0];
  const float* wf   = (const float*)d_in[1];
  const float* hop1 = (const float*)d_in[2];
  const float* hop2 = (const float*)d_in[3];
  const float* hop3 = (const float*)d_in[4];
  const float* W1   = (const float*)d_in[5];
  const float* b1   = (const float*)d_in[6];
  const float* W2   = (const float*)d_in[7];
  const float* b2   = (const float*)d_in[8];
  float* out = (float*)d_out;

  // workspace layout
  char* ws = (char*)d_ws;
  unsigned short* hops0 = (unsigned short*)(ws);                 // 1 MB bf16
  unsigned short* Xt    = (unsigned short*)(ws + (1u << 20));    // 1 MB bf16
  float* hopsP          = (float*)(ws + (2u << 20));             // 12 MB f32
  unsigned short* W1t   = (unsigned short*)(ws + (14u << 20));   // 64 KB
  unsigned short* W2t   = (unsigned short*)(ws + (14u << 20) + 65536);

  prep_x<<<(NROWS * 64) / 256, 256, 0, stream>>>(x, wf, Xt, hops0);
  prep_w<<<32768 / 256, 256, 0, stream>>>(W1, W2, W1t, W2t);
  hop_gemm<<<384, 256, 0, stream>>>(hop1, hop2, hop3, Xt, hopsP);
  mlp<<<64, 256, 0, stream>>>(hops0, hopsP, W1t, b1, W2t, b2, out);
}
